// SparseMOE_52836687675416
// MI455X (gfx1250) — matrix operand order repaired and checked by reading, not compiled, not run on hardware
//
#include <hip/hip_runtime.h>
#include <hip/hip_fp16.h>

typedef __attribute__((ext_vector_type(16))) _Float16 v16h;
typedef __attribute__((ext_vector_type(8)))  _Float16 h8;
typedef __attribute__((ext_vector_type(8)))  float    v8f;
typedef __attribute__((ext_vector_type(4)))  float    f4;

#define DDIM 1024
#define DFF  2048
#define NE   8
#define NTOK 4096
#define MT   64     // token rows per block
#define NT   128    // output cols per block
#define KT   32     // K step (f16 WMMA K)
#define LDP  (KT + 8)   // padded LDS row stride (40 halves = 80B: 16B-aligned, conflict-free)

// ---------------------------------------------------------------- zero + counts
__global__ __launch_bounds__(256) void moe_zero(float* __restrict__ out, int* __restrict__ counts) {
    size_t i  = (size_t)blockIdx.x * blockDim.x + threadIdx.x;
    size_t n4 = (size_t)NTOK * DDIM / 4;
    f4 z = {0.f, 0.f, 0.f, 0.f};
    for (size_t j = i; j < n4; j += (size_t)gridDim.x * blockDim.x)
        ((f4*)out)[j] = z;
    if (blockIdx.x == 0 && threadIdx.x < NE) counts[threadIdx.x] = 0;
}

// ---------------------------------------------------------------- router (fp32 exact)
__global__ __launch_bounds__(256) void moe_router(const float* __restrict__ x,
                                                  const float* __restrict__ gw,
                                                  const float* __restrict__ gb,
                                                  float* __restrict__ logits_out,
                                                  int* __restrict__ counts,
                                                  int* __restrict__ bucket,
                                                  float* __restrict__ bw) {
    __shared__ float xs[DDIM];
    __shared__ float lg[NE];
    int t = blockIdx.x;
    const float* xt = x + (size_t)t * DDIM;
    for (int i = threadIdx.x; i < DDIM; i += 256) xs[i] = xt[i];
    __syncthreads();

    int wave = threadIdx.x >> 5;   // 8 waves == 8 experts
    int lane = threadIdx.x & 31;
    {
        const float* w = gw + (size_t)wave * DDIM;
        float s = 0.f;
        for (int i = lane; i < DDIM; i += 32) s += xs[i] * w[i];
        for (int off = 16; off > 0; off >>= 1) s += __shfl_xor(s, off, 32);
        if (lane == 0) lg[wave] = s + gb[wave];
    }
    __syncthreads();

    if (threadIdx.x == 0) {
        float l[NE];
        #pragma unroll
        for (int e = 0; e < NE; ++e) {
            l[e] = lg[e];
            logits_out[(size_t)t * NE + e] = l[e];
        }
        int i1 = 0;
        #pragma unroll
        for (int e = 1; e < NE; ++e) if (l[e] > l[i1]) i1 = e;   // first max on ties
        int i2 = (i1 == 0) ? 1 : 0;
        #pragma unroll
        for (int e = 0; e < NE; ++e) if (e != i1 && l[e] > l[i2]) i2 = e;
        // renormalized top-2 softmax weights: exp(l1)/(exp(l1)+exp(l2)) (exact div: output-critical)
        float w1 = 1.f / (1.f + __expf(l[i2] - l[i1]));
        float w2 = 1.f - w1;
        int p1 = atomicAdd(&counts[i1], 1);
        bucket[i1 * NTOK + p1] = t;  bw[i1 * NTOK + p1] = w1;
        int p2 = atomicAdd(&counts[i2], 1);
        bucket[i2 * NTOK + p2] = t;  bw[i2 * NTOK + p2] = w2;
    }
}

// ---------------------------------------------------------------- tiny scan
__global__ void moe_prefix(const int* __restrict__ counts, int* __restrict__ offsets) {
    if (threadIdx.x == 0 && blockIdx.x == 0) {
        int acc = 0;
        for (int e = 0; e < NE; ++e) { offsets[e] = acc; acc += counts[e]; }
        offsets[NE] = acc;
    }
}

// helpers: build WMMA fragments from padded LDS tiles
__device__ __forceinline__ v16h load_fragA(const _Float16* base, int lane) {
    // A 16x32 f16 layout: lanes 0-15 -> K[0..7]|K[16..23]; lanes 16-31 -> K[8..15]|K[24..31]
    int kb = (lane >> 4) << 3;
    h8 lo = *(const h8*)(base + kb);
    h8 hi = *(const h8*)(base + 16 + kb);
    v16h a;
    #pragma unroll
    for (int i = 0; i < 8; ++i) { a[i] = lo[i]; a[8 + i] = hi[i]; }
    return a;
}
__device__ __forceinline__ v16h load_fragB(const _Float16* base, int lane) {
    // B 32x16 f16 layout: lanes 0-15 -> K[0..7]|K[16..23]; lanes 16-31 -> K[8..15]|K[24..31]
    int kb = (lane >> 4) << 3;
    h8 lo = *(const h8*)(base + kb);
    h8 hi = *(const h8*)(base + 16 + kb);
    v16h b;
    #pragma unroll
    for (int i = 0; i < 8; ++i) { b[i] = lo[i]; b[8 + i] = hi[i]; }
    return b;
}

// ---------------------------------------------------------------- GEMM1: act = w * silu(x@Wx) * (x@Vx)
__global__ __launch_bounds__(256) void moe_ffn1(const float* __restrict__ x,
                                                const float* __restrict__ Wx,
                                                const float* __restrict__ Vx,
                                                const int* __restrict__ counts,
                                                const int* __restrict__ offsets,
                                                const int* __restrict__ bucket,
                                                const float* __restrict__ bw,
                                                _Float16* __restrict__ act) {
    int e   = blockIdx.z;
    int cnt = counts[e];
    int m0  = blockIdx.x * MT;
    if (m0 >= cnt) return;
    int n0  = blockIdx.y * NT;

    __shared__ _Float16 As [MT][LDP];
    __shared__ _Float16 BsW[NT][LDP];
    __shared__ _Float16 BsV[NT][LDP];
    __shared__ int   toks[MT];
    __shared__ float wts [MT];

    for (int i = threadIdx.x; i < MT; i += 256) {
        int m = m0 + i;
        toks[i] = (m < cnt) ? bucket[e * NTOK + m] : 0;
        wts[i]  = (m < cnt) ? bw[e * NTOK + m]     : 0.f;
    }

    int wave = threadIdx.x >> 5, lane = threadIdx.x & 31;
    int wm = wave & 3;        // row subtile 0..3  (16 rows each)
    int wn = wave >> 2;       // col half 0..1     (64 cols each)
    int row = lane & 15;

    v8f accH[4], accV[4];
    #pragma unroll
    for (int c = 0; c < 4; ++c) { accH[c] = (v8f){}; accV[c] = (v8f){}; }

    const float* WxE = Wx + (size_t)e * DDIM * DFF;
    const float* VxE = Vx + (size_t)e * DDIM * DFF;

    for (int k0 = 0; k0 < DDIM; k0 += KT) {
        __syncthreads();
        // A tile: 64 tokens x 32 K (gathered, f32 -> f16)
        for (int i = threadIdx.x; i < MT * KT; i += 256) {
            int m = i >> 5, kk = i & 31;
            float v = ((m0 + m) < cnt) ? x[(size_t)toks[m] * DDIM + k0 + kk] : 0.f;
            As[m][kk] = (_Float16)v;
        }
        // B tiles: Wx & Vx, 32 K x 128 N, stored [n][k] (transposed, f32 -> f16)
        for (int i = threadIdx.x; i < KT * NT; i += 256) {
            int kk = i >> 7, nn = i & 127;
            size_t g = (size_t)(k0 + kk) * DFF + (n0 + nn);
            BsW[nn][kk] = (_Float16)WxE[g];
            BsV[nn][kk] = (_Float16)VxE[g];
        }
        __syncthreads();

        v16h a = load_fragA(&As[wm * 16 + row][0], lane);
        #pragma unroll
        for (int c = 0; c < 4; ++c) {
            int ncol = wn * 64 + c * 16 + row;
            v16h b = load_fragB(&BsW[ncol][0], lane);
            accH[c] = __builtin_amdgcn_wmma_f32_16x16x32_f16(false, a, false, b,
                                                             (short)0, accH[c], false, false);
            b = load_fragB(&BsV[ncol][0], lane);
            accV[c] = __builtin_amdgcn_wmma_f32_16x16x32_f16(false, a, false, b,
                                                             (short)0, accV[c], false, false);
        }
    }

    // epilogue: act[compact_row][col] = w_token * silu(h) * v   (f16)
    // silu via v_rcp_f32: ~1ulp rcp error is far below the f16 storage rounding.
    int off = offsets[e];
    #pragma unroll
    for (int c = 0; c < 4; ++c) {
        int col = n0 + wn * 64 + c * 16 + (lane & 15);
        #pragma unroll
        for (int r = 0; r < 8; ++r) {
            int mloc = wm * 16 + ((lane >> 4) << 3) + r;   // C layout: lanes 16-31 -> M+8
            int m = m0 + mloc;
            if (m < cnt) {
                float h = accH[c][r], vv = accV[c][r];
                float sig  = __builtin_amdgcn_rcpf(1.f + __expf(-h));
                float aval = h * sig * vv * wts[mloc];
                act[(size_t)(off + m) * DFF + col] = (_Float16)aval;
            }
        }
    }
}

// ---------------------------------------------------------------- GEMM2: out += act @ Wo (scatter-add)
__global__ __launch_bounds__(256) void moe_ffn2(const _Float16* __restrict__ act,
                                                const float* __restrict__ Wo,
                                                const int* __restrict__ counts,
                                                const int* __restrict__ offsets,
                                                const int* __restrict__ bucket,
                                                float* __restrict__ out) {
    int e   = blockIdx.z;
    int cnt = counts[e];
    int m0  = blockIdx.x * MT;
    if (m0 >= cnt) return;
    int n0  = blockIdx.y * NT;

    __shared__ _Float16 As[MT][LDP];
    __shared__ _Float16 Bs[NT][LDP];

    int off  = offsets[e];
    int wave = threadIdx.x >> 5, lane = threadIdx.x & 31;
    int wm = wave & 3, wn = wave >> 2;
    int row = lane & 15;

    v8f acc[4];
    #pragma unroll
    for (int c = 0; c < 4; ++c) acc[c] = (v8f){};

    const float* WoE = Wo + (size_t)e * DFF * DDIM;
    const h8 hz = {};

    for (int k0 = 0; k0 < DFF; k0 += KT) {
        __syncthreads();
        // A tile: already f16 in ws; 64 x 32, one b128 per thread
        for (int i = threadIdx.x; i < MT * KT / 8; i += 256) {
            int m = i >> 2, k8 = (i & 3) << 3;
            int mabs = m0 + m;
            h8 v = (mabs < cnt)
                 ? *(const h8*)&act[(size_t)(off + mabs) * DFF + k0 + k8]
                 : hz;
            *(h8*)&As[m][k8] = v;
        }
        // B tile: Wo 32 K x 128 N, stored [n][k] (f32 -> f16)
        for (int i = threadIdx.x; i < KT * NT; i += 256) {
            int kk = i >> 7, nn = i & 127;
            Bs[nn][kk] = (_Float16)WoE[(size_t)(k0 + kk) * DDIM + (n0 + nn)];
        }
        __syncthreads();

        v16h a = load_fragA(&As[wm * 16 + row][0], lane);
        #pragma unroll
        for (int c = 0; c < 4; ++c) {
            int ncol = wn * 64 + c * 16 + row;
            v16h b = load_fragB(&Bs[ncol][0], lane);
            acc[c] = __builtin_amdgcn_wmma_f32_16x16x32_f16(false, a, false, b,
                                                            (short)0, acc[c], false, false);
        }
    }

    #pragma unroll
    for (int c = 0; c < 4; ++c) {
        int col = n0 + wn * 64 + c * 16 + (lane & 15);
        #pragma unroll
        for (int r = 0; r < 8; ++r) {
            int mloc = wm * 16 + ((lane >> 4) << 3) + r;
            int m = m0 + mloc;
            if (m < cnt) {
                int t = bucket[e * NTOK + m];
                atomicAdd(&out[(size_t)t * DDIM + col], acc[c][r]);
            }
        }
    }
}

// ---------------------------------------------------------------- launch
extern "C" void kernel_launch(void* const* d_in, const int* in_sizes, int n_in,
                              void* d_out, int out_size, void* d_ws, size_t ws_size,
                              hipStream_t stream) {
    const float* x      = (const float*)d_in[0];
    const float* gate_w = (const float*)d_in[1];
    const float* gate_b = (const float*)d_in[2];
    const float* Wx     = (const float*)d_in[3];
    const float* Vx     = (const float*)d_in[4];
    const float* Wo     = (const float*)d_in[5];

    float* out    = (float*)d_out;                       // [T, D]
    float* logits = out + (size_t)NTOK * DDIM;           // [T, E] appended

    // workspace layout
    char* ws = (char*)d_ws;
    int*      counts  = (int*)ws;                                  // 8 ints
    int*      offsets = (int*)(ws + 64);                           // 9 ints
    int*      bucket  = (int*)(ws + 256);                          // E*T ints
    float*    bw      = (float*)(ws + 256 + (size_t)NE * NTOK * 4);// E*T floats
    _Float16* act     = (_Float16*)(ws + 256 + (size_t)NE * NTOK * 8); // 2T x DFF f16 (~32MB)

    moe_zero  <<<2048, 256, 0, stream>>>(out, counts);
    moe_router<<<NTOK, 256, 0, stream>>>(x, gate_w, gate_b, logits, counts, bucket, bw);
    moe_prefix<<<1, 32, 0, stream>>>(counts, offsets);
    moe_ffn1  <<<dim3(NTOK / MT, DFF / NT, NE), 256, 0, stream>>>(x, Wx, Vx, counts, offsets,
                                                                  bucket, bw, act);
    moe_ffn2  <<<dim3(NTOK / MT, DDIM / NT, NE), 256, 0, stream>>>(act, Wo, counts, offsets,
                                                                   bucket, out);
}
